// GDN_63702954934564
// MI455X (gfx1250) — matrix operand order repaired and checked by reading, not compiled, not run on hardware
//
#include <hip/hip_runtime.h>
#include <hip/hip_bf16.h>

#define NN 2048
#define TT 5
#define DD 64
#define TOPKK 21
#define BB 32
#define EPSF 1e-5f
#define NEG_SLOPE 0.2f
#define TK_WAVES 4

typedef __attribute__((ext_vector_type(16))) _Float16 v16h;
typedef __attribute__((ext_vector_type(8)))  float    v8f;
typedef int async_i4 __attribute__((vector_size(16)));   // matches builtin's V4i pointee

#if defined(__gfx1250__) && __has_builtin(__builtin_amdgcn_global_load_async_to_lds_b128) && __has_builtin(__builtin_amdgcn_s_wait_asynccnt)
#define HAVE_ASYNC_LDS 1
#else
#define HAVE_ASYNC_LDS 0
#endif

// ---------------- Kernel 1: normalize emb -> f16, emb-half attention dots ---
__global__ __launch_bounds__(64)
void norm_kernel(const float* __restrict__ emb, const float* __restrict__ a_src,
                 const float* __restrict__ a_dst, _Float16* __restrict__ ehat,
                 float* __restrict__ es_emb, float* __restrict__ ed_emb) {
  const int n = blockIdx.x;
  const int d = threadIdx.x;
  __shared__ float red[64];
  const float v = emb[(size_t)n * DD + d];

  red[d] = v * v;
  __syncthreads();
  for (int s = 32; s > 0; s >>= 1) { if (d < s) red[d] += red[d + s]; __syncthreads(); }
  const float rn = rsqrtf(red[0]);
  __syncthreads();
  ehat[(size_t)n * DD + d] = (_Float16)(v * rn);

  red[d] = v * a_src[DD + d];
  __syncthreads();
  for (int s = 32; s > 0; s >>= 1) { if (d < s) red[d] += red[d + s]; __syncthreads(); }
  if (d == 0) es_emb[n] = red[0];
  __syncthreads();

  red[d] = v * a_dst[DD + d];
  __syncthreads();
  for (int s = 32; s > 0; s >>= 1) { if (d < s) red[d] += red[d + s]; __syncthreads(); }
  if (d == 0) ed_emb[n] = red[0];
}

// ---------------- Kernel 2: cos = ehat @ ehat^T via WMMA (f16 in, f32 acc) --
// One wave computes a 16x64 strip: A fragments loaded once, reused across
// 4 j-tiles (2 chained v_wmma_f32_16x16x32_f16 per tile, K=64).
__global__ __launch_bounds__(32)
void cos_wmma_kernel(const _Float16* __restrict__ ehat, float* __restrict__ cosm) {
  const int i0 = blockIdx.y * 16;
  const int j0base = blockIdx.x * 64;
  const int lane = threadIdx.x;          // 0..31, EXEC all ones
  const int rsel = lane & 15;
  const _Float16* arow = ehat + (size_t)(i0 + rsel) * DD;  // A: row of tile i
  const int abase = (lane & 16) ? 8 : 0;   // A 16-bit layout: hi-half lanes hold K+8
  const int kb    = (lane & 16) ? 16 : 0;  // unused: B is loaded with abase, same k order as A
  const int mb    = (lane & 16) ? 8 : 0;   // C layout: VGPR v holds M = mb + v

  v16h a0, a1;
#pragma unroll
  for (int i = 0; i < 16; ++i) {
    const int r = i >> 1;
    const int k = (r < 4) ? (abase + 2 * r + (i & 1))
                          : (16 + abase + 2 * (r - 4) + (i & 1));
    a0[i] = arow[k];
    a1[i] = arow[k + 32];
  }

#pragma unroll
  for (int jt = 0; jt < 4; ++jt) {
    const int j0 = j0base + jt * 16;
    const _Float16* brow = ehat + (size_t)(j0 + rsel) * DD;  // col n = row j0+n
    v16h b0, b1;
#pragma unroll
    for (int i = 0; i < 16; ++i) {
      b0[i] = brow[abase + i + (i & 8)];
      b1[i] = brow[abase + i + (i & 8) + 32];
    }
    v8f c = {};
    c = __builtin_amdgcn_wmma_f32_16x16x32_f16(false, a0, false, b0, (short)0, c, false, false);
    c = __builtin_amdgcn_wmma_f32_16x16x32_f16(false, a1, false, b1, (short)0, c, false, false);
#pragma unroll
    for (int v = 0; v < 8; ++v)
      cosm[(size_t)(i0 + mb + v) * NN + (j0 + rsel)] = c[v];
  }
}

// ---------------- Kernel 3: top-21 per row (one wave per row, LDS-staged) ---
// Stage each 8KB row into LDS once (async memory->LDS on CDNA5: no VGPR
// round-trip, ASYNCcnt-tracked), then run the 21 knockout argmax passes out
// of LDS instead of re-reading 21 x 16MB from L2.
__global__ __launch_bounds__(128)
void topk_kernel(const float* __restrict__ cosm, int* __restrict__ idx) {
  __shared__ __align__(16) float rows[TK_WAVES][NN];
  const int lane = threadIdx.x & 31;
  const int wave = threadIdx.x >> 5;
  const int n = blockIdx.x * TK_WAVES + wave;
  const float* row = cosm + (size_t)n * NN;
  float* lrow = rows[wave];

#if HAVE_ASYNC_LDS
#pragma unroll
  for (int t = 0; t < NN / (32 * 4); ++t) {        // 16 x b128 per lane
    const int c = (t * 32 + lane) * 4;
    __builtin_amdgcn_global_load_async_to_lds_b128(
        (__attribute__((address_space(1))) async_i4*)(row + c),
        (__attribute__((address_space(3))) async_i4*)(lrow + c), 0, 0);
  }
  __builtin_amdgcn_s_wait_asynccnt(0);
#else
  for (int c = lane; c < NN / 4; c += 32)
    ((float4*)lrow)[c] = ((const float4*)row)[c];
#endif
  // wave-private row: no cross-wave sync needed

  for (int k = 0; k < TOPKK; ++k) {
    float best = -3.0e38f;
    int bidx = NN;
    for (int c = lane; c < NN; c += 32) {
      const float v = lrow[c];
      if (v > best) { best = v; bidx = c; }
    }
#pragma unroll
    for (int off = 16; off > 0; off >>= 1) {
      const float ov = __shfl_xor(best, off, 32);
      const int   oi = __shfl_xor(bidx, off, 32);
      if (ov > best || (ov == best && oi < bidx)) { best = ov; bidx = oi; }
    }
    if (lane == 0) idx[n * TOPKK + k] = bidx;
    // knockout in LDS by the owning lane (same-lane reload -> ordered)
    if ((bidx & 31) == lane) lrow[bidx] = -3.0e38f;
  }
}

// ---------------- Kernel 4: h = x W^T + b ; attention logits ---------------
__global__ __launch_bounds__(64)
void h_e_kernel(const float* __restrict__ x, const float* __restrict__ W,
                const float* __restrict__ Wb, const float* __restrict__ a_src,
                const float* __restrict__ a_dst, const float* __restrict__ es_emb,
                const float* __restrict__ ed_emb, float* __restrict__ h,
                float* __restrict__ e_src, float* __restrict__ e_dst) {
  const int bn = blockIdx.x;
  const int d  = threadIdx.x;
  const int n  = bn & (NN - 1);
  __shared__ float xs[TT];
  __shared__ float red[64];
  if (d < TT) xs[d] = x[(size_t)bn * TT + d];
  __syncthreads();
  float acc = Wb[d];
#pragma unroll
  for (int t = 0; t < TT; ++t) acc = fmaf(xs[t], W[d * TT + t], acc);
  h[(size_t)bn * DD + d] = acc;

  red[d] = acc * a_src[d];
  __syncthreads();
  for (int s = 32; s > 0; s >>= 1) { if (d < s) red[d] += red[d + s]; __syncthreads(); }
  if (d == 0) e_src[bn] = red[0] + es_emb[n];
  __syncthreads();

  red[d] = acc * a_dst[d];
  __syncthreads();
  for (int s = 32; s > 0; s >>= 1) { if (d < s) red[d] += red[d + s]; __syncthreads(); }
  if (d == 0) e_dst[bn] = red[0] + ed_emb[n];
}

// ---------------- Kernel 5: sparse softmax + aggregate + z*emb -------------
__global__ __launch_bounds__(64)
void attn_kernel(const int* __restrict__ idx, const float* __restrict__ e_src,
                 const float* __restrict__ e_dst, const float* __restrict__ h,
                 const float* __restrict__ emb, float* __restrict__ outbuf) {
  const int i = blockIdx.x;
  const int b = blockIdx.y;
  const int d = threadIdx.x;
  __shared__ float alpha[TOPKK];
  __shared__ int   nb[TOPKK];
  const float* hb = h + (size_t)b * NN * DD;

  if (d < TOPKK) {
    const int j = idx[i * TOPKK + d];
    nb[d] = j;
    float e = e_src[b * NN + i] + e_dst[b * NN + j];
    alpha[d] = (e > 0.0f) ? e : NEG_SLOPE * e;     // leaky_relu before masking
    __builtin_prefetch(hb + (size_t)j * DD, 0, 0); // global_prefetch_b8 neighbor rows
  }
  __syncthreads();
  if (d == 0) {
    float m = -3.0e38f;
    for (int k = 0; k < TOPKK; ++k) m = fmaxf(m, alpha[k]);
    float s = 0.0f;
    for (int k = 0; k < TOPKK; ++k) { const float ev = __expf(alpha[k] - m); alpha[k] = ev; s += ev; }
    const float inv = 1.0f / s;
    for (int k = 0; k < TOPKK; ++k) alpha[k] *= inv;
  }
  __syncthreads();
  float z = 0.0f;
#pragma unroll
  for (int k = 0; k < TOPKK; ++k) z = fmaf(alpha[k], hb[(size_t)nb[k] * DD + d], z);
  z = fmaxf(z, 0.0f);
  outbuf[((size_t)b * NN + i) * DD + d] = z * emb[(size_t)i * DD + d];
}

// ---------------- Kernel 6: BatchNorm stats over (B,N) per channel ---------
__global__ __launch_bounds__(256)
void bn_stats_kernel(const float* __restrict__ outbuf, float* __restrict__ mean,
                     float* __restrict__ rstd) {
  const int d = blockIdx.x;
  const int t = threadIdx.x;
  __shared__ float s1[256], s2[256];
  float a = 0.0f, q = 0.0f;
  for (int r = t; r < BB * NN; r += 256) {
    const float v = outbuf[(size_t)r * DD + d];
    a += v; q = fmaf(v, v, q);
  }
  s1[t] = a; s2[t] = q;
  __syncthreads();
  for (int s = 128; s > 0; s >>= 1) {
    if (t < s) { s1[t] += s1[t + s]; s2[t] += s2[t + s]; }
    __syncthreads();
  }
  if (t == 0) {
    const float inv_n = 1.0f / (float)(BB * NN);
    const float mu = s1[0] * inv_n;
    const float var = s2[0] * inv_n - mu * mu;
    mean[d] = mu;
    rstd[d] = rsqrtf(var + EPSF);
  }
}

// ---------------- Kernel 7: BN + relu + fc ---------------------------------
__global__ __launch_bounds__(64)
void head_kernel(const float* __restrict__ outbuf, const float* __restrict__ mean,
                 const float* __restrict__ rstd, const float* __restrict__ gamma,
                 const float* __restrict__ beta, const float* __restrict__ fc_w,
                 const float* __restrict__ fc_b, float* __restrict__ y) {
  const int bn = blockIdx.x;
  const int d  = threadIdx.x;
  __shared__ float red[64];
  float v = (outbuf[(size_t)bn * DD + d] - mean[d]) * rstd[d] * gamma[d] + beta[d];
  v = fmaxf(v, 0.0f);
  red[d] = v * fc_w[d];
  __syncthreads();
  for (int s = 32; s > 0; s >>= 1) { if (d < s) red[d] += red[d + s]; __syncthreads(); }
  if (d == 0) y[bn] = red[0] + fc_b[0];
}

extern "C" void kernel_launch(void* const* d_in, const int* in_sizes, int n_in,
                              void* d_out, int out_size, void* d_ws, size_t ws_size,
                              hipStream_t stream) {
  (void)in_sizes; (void)n_in; (void)out_size; (void)ws_size;
  const float* x      = (const float*)d_in[0];
  const float* emb    = (const float*)d_in[1];
  const float* W      = (const float*)d_in[2];
  const float* Wb     = (const float*)d_in[3];
  const float* a_src  = (const float*)d_in[4];
  const float* a_dst  = (const float*)d_in[5];
  const float* gamma  = (const float*)d_in[6];
  const float* beta   = (const float*)d_in[7];
  const float* fc_w   = (const float*)d_in[8];
  const float* fc_b   = (const float*)d_in[9];
  float* y = (float*)d_out;

  char* ws = (char*)d_ws;
  size_t off = 0;
  auto alloc = [&](size_t bytes) -> char* {
    char* p = ws + off;
    off += (bytes + 255) & ~(size_t)255;
    return p;
  };
  float*    cosm   = (float*)alloc((size_t)NN * NN * sizeof(float));      // 16 MB
  float*    h      = (float*)alloc((size_t)BB * NN * DD * sizeof(float)); // 16 MB
  _Float16* ehat   = (_Float16*)alloc((size_t)NN * DD * sizeof(_Float16));
  float*    es_emb = (float*)alloc((size_t)NN * sizeof(float));
  float*    ed_emb = (float*)alloc((size_t)NN * sizeof(float));
  float*    e_src  = (float*)alloc((size_t)BB * NN * sizeof(float));
  float*    e_dst  = (float*)alloc((size_t)BB * NN * sizeof(float));
  int*      idx    = (int*)alloc((size_t)NN * TOPKK * sizeof(int));
  float*    mean   = (float*)alloc((size_t)DD * sizeof(float));
  float*    rstd   = (float*)alloc((size_t)DD * sizeof(float));
  float*    outbuf = cosm;  // alias: cos is fully consumed by topk before kernel 5 writes

  norm_kernel<<<NN, 64, 0, stream>>>(emb, a_src, a_dst, ehat, es_emb, ed_emb);
  cos_wmma_kernel<<<dim3(NN / 64, NN / 16), 32, 0, stream>>>(ehat, cosm);
  topk_kernel<<<NN / TK_WAVES, 128, 0, stream>>>(cosm, idx);
  h_e_kernel<<<BB * NN, 64, 0, stream>>>(x, W, Wb, a_src, a_dst, es_emb, ed_emb,
                                         h, e_src, e_dst);
  attn_kernel<<<dim3(NN, BB), 64, 0, stream>>>(idx, e_src, e_dst, h, emb, outbuf);
  bn_stats_kernel<<<DD, 256, 0, stream>>>(outbuf, mean, rstd);
  head_kernel<<<BB * NN, 64, 0, stream>>>(outbuf, mean, rstd, gamma, beta,
                                          fc_w, fc_b, y);
}
